// GraphAttention_65876208386282
// MI455X (gfx1250) — matrix operand order repaired and checked by reading, not compiled, not run on hardware
//
#include <hip/hip_runtime.h>
#include <math.h>

typedef __attribute__((ext_vector_type(16))) _Float16 v16h;
typedef __attribute__((ext_vector_type(8)))  _Float16 v8h;
typedef __attribute__((ext_vector_type(4)))  _Float16 v4h;
typedef __attribute__((ext_vector_type(8)))  float    v8f;

constexpr int NN = 10000, NE = 160000, RROWS = NN * 9;

// f16 weight pool (offsets in halfs) -- stored PRE-SWIZZLED in WMMA B-fragment order:
// frag (nt, ks, lane) occupies halfs [((nt*2+ks)*32+lane)*16, +16) and is loaded as one v16h.
constexpr int WSD_H = 0;       // [Wsrc|Wdst] 64x128
constexpr int W1_H  = 8192;    // rad_W1 64x64
constexpr int W2_H  = 12288;   // rad_W2 64x192
constexpr int WA_H  = 24576;   // W_alpha 64x64
constexpr int WV_H  = 28672;   // W_value 64x128
constexpr int WP_H  = 36864;   // W_proj 64x64

// workspace byte offsets
constexpr size_t OFF_WH   = 0;
constexpr size_t OFF_LIN  = 128 * 1024;                              // 90000x128 f32
constexpr size_t OFF_WLC  = OFF_LIN  + (size_t)RROWS * 128 * 4;      // 160000x192 f32
constexpr size_t OFF_LOG  = OFF_WLC  + (size_t)NE * 192 * 4;         // 160000x8 f32
constexpr size_t OFF_NMAX = OFF_LOG  + (size_t)NE * 8 * 4;           // 10000x8 u32
constexpr size_t OFF_NSUM = OFF_NMAX + (size_t)NN * 8 * 4;           // 10000x8 f32
constexpr size_t OFF_ACC  = OFF_NSUM + (size_t)NN * 8 * 4;           // 90000x64 f32

// ---------------- WMMA fragment helpers (CDNA5 16x16x32 f16, wave32) ----------------
__device__ inline v16h frag_a_lds(const _Float16* A /*16x64 row-major*/, int lane, int k0) {
  int hf = lane >> 4, m = lane & 15;
  const _Float16* row = A + m * 64;
  v8h lo = *(const v8h*)(row + k0 + 8 * hf);
  v8h hi = *(const v8h*)(row + k0 + 16 + 8 * hf);
  v16h a;
#pragma unroll
  for (int j = 0; j < 8; ++j) { a[j] = lo[j]; a[8 + j] = hi[j]; }
  return a;
}
__device__ inline v16h frag_b_sw(const _Float16* Bsw, int nt, int ks, int lane) {
  return *(const v16h*)(Bsw + ((((nt << 1) + ks) << 5) + lane) * 16);
}
__device__ inline v8f wmma32(v16h a, v16h b, v8f c) {
  return __builtin_amdgcn_wmma_f32_16x16x32_f16(false, a, false, b, (short)0, c, false, false);
}

__device__ inline float sigmf(float x) { return 1.0f / (1.0f + __expf(-x)); }
__device__ inline float siluf(float x) { return x * sigmf(x); }
__device__ inline float slrelu(float x) { return 0.6f * x + 0.4f * x * (2.0f * sigmf(x) - 1.0f); }
__device__ inline unsigned ford(float x) {
  unsigned u = __float_as_uint(x);
  return (u & 0x80000000u) ? ~u : (u | 0x80000000u);
}
__device__ inline float funord(unsigned u) {
  return (u & 0x80000000u) ? __uint_as_float(u & 0x7FFFFFFFu) : __uint_as_float(~u);
}
// stage 16x64 f32 rows -> f16 LDS tile, vectorized (float4 x2 -> v8h)
__device__ inline void stage_a16x64(const float* Xr, _Float16* A, int lane) {
#pragma unroll
  for (int it = 0; it < 4; ++it) {
    int base = (it * 32 + lane) * 8;
    float4 x0 = *(const float4*)(Xr + base);
    float4 x1 = *(const float4*)(Xr + base + 4);
    v8h h;
    h[0] = (_Float16)x0.x; h[1] = (_Float16)x0.y; h[2] = (_Float16)x0.z; h[3] = (_Float16)x0.w;
    h[4] = (_Float16)x1.x; h[5] = (_Float16)x1.y; h[6] = (_Float16)x1.z; h[7] = (_Float16)x1.w;
    *(v8h*)(A + base) = h;
  }
}

// ---------------- K0: convert weights to f16 in B-fragment-swizzled order ----------------
__device__ inline void swz_idx(int i, int& k, int& nc) {
  int fragidx = i >> 4; int j = i & 15;
  int lane = fragidx & 31, ks = (fragidx >> 5) & 1, nt = fragidx >> 6;
  int hf = lane >> 4, n = lane & 15;
  k  = ks * 32 + (j < 8 ? 8 * hf + j : 16 + 8 * hf + (j - 8));
  nc = nt * 16 + n;
}
__global__ void k_convert_weights(const float* Wsrc, const float* Wdst, const float* W1,
                                  const float* W2, const float* Wa, const float* Wv,
                                  const float* Wp, _Float16* wh) {
  int t = blockIdx.x * blockDim.x + threadIdx.x;
  int stride = gridDim.x * blockDim.x;
  int k, c;
  for (int i = t; i < 64 * 128; i += stride) {
    swz_idx(i, k, c);
    wh[WSD_H + i] = (_Float16)(c < 64 ? Wsrc[k * 64 + c] : Wdst[k * 64 + (c - 64)]);
  }
  for (int i = t; i < 64 * 64;  i += stride) { swz_idx(i, k, c); wh[W1_H + i] = (_Float16)W1[k * 64 + c]; }
  for (int i = t; i < 64 * 192; i += stride) { swz_idx(i, k, c); wh[W2_H + i] = (_Float16)W2[k * 192 + c]; }
  for (int i = t; i < 64 * 64;  i += stride) { swz_idx(i, k, c); wh[WA_H + i] = (_Float16)Wa[k * 64 + c]; }
  for (int i = t; i < 64 * 128; i += stride) { swz_idx(i, k, c); wh[WV_H + i] = (_Float16)Wv[k * 128 + c]; }
  for (int i = t; i < 64 * 64;  i += stride) { swz_idx(i, k, c); wh[WP_H + i] = (_Float16)Wp[k * 64 + c]; }
}

// ---------------- K1: fused node linear (90000x64)@(64x128) -> lin (no bias) ----------------
__global__ void __launch_bounds__(256) k_node_lin(const float* X, const _Float16* wh,
                                                  float* lin) {
  __shared__ _Float16 As[8][1024];
  int wid = threadIdx.x >> 5, lane = threadIdx.x & 31;
  int tile = blockIdx.x * 8 + wid;
  bool active = tile * 16 < RROWS;
  int row0 = tile * 16;
  _Float16* A = As[wid];
  if (active) stage_a16x64(X + (size_t)row0 * 64, A, lane);
  __syncthreads();
  int hf = lane >> 4, n = lane & 15;
  for (int nt = 0; nt < 8; ++nt) {
    v8f acc = {};
#pragma unroll
    for (int ks = 0; ks < 2; ++ks)
      acc = wmma32(frag_a_lds(A, lane, ks * 32), frag_b_sw(wh + WSD_H, nt, ks, lane), acc);
    if (active) {
      int col = nt * 16 + n;
      float* outp = lin + (size_t)(row0 + (hf ? 8 : 0)) * 128 + col;
#pragma unroll
      for (int r = 0; r < 8; ++r) outp[r * 128] = acc[r];
    }
  }
}

// ---------------- K1b/K5b: add bias to the m==0 rows (rows n*9) ----------------
__global__ void k_rowbias(float* buf, const float* bias, int ld) {
  int t = blockIdx.x * blockDim.x + threadIdx.x;
  if (t >= NN * 64) return;
  int nnode = t >> 6, c = t & 63;
  buf[(size_t)nnode * 9 * ld + c] += bias[c];
}

// ---------------- K2: radial MLP + LN + SiLU -> wlc ; msg(m=0) -> alpha logits ----------------
__global__ void __launch_bounds__(256) k_radial_alpha(
    const float* escal, const float* b1, const float* ln_g, const float* ln_b,
    const float* rad_off, const float* b_alpha, const float* alpha_dot,
    const float* edge_attr, const int* esrc, const int* edst,
    const float* lin, const _Float16* wh, float* wlc, float* logits) {
  __shared__ _Float16 As[8][1024];
  __shared__ float    Hs[8][1024];
  __shared__ int      SI[8][16];
  __shared__ float    EA[8][16];
  __shared__ int      DI[8][16];
  __shared__ float    ADOT[64];
  int wid = threadIdx.x >> 5, lane = threadIdx.x & 31;
  int tile = blockIdx.x * 8 + wid;      // 10000 tiles exactly
  int e0 = tile * 16;
  _Float16* A = As[wid];
  float* Hw = Hs[wid];
  if (threadIdx.x < 64) ADOT[threadIdx.x] = alpha_dot[threadIdx.x];
  if (lane < 16) {
    SI[wid][lane] = esrc[e0 + lane];
    DI[wid][lane] = edst[e0 + lane];
    EA[wid][lane] = edge_attr[(e0 + lane) * 9];
  }
  stage_a16x64(escal + (size_t)e0 * 64, A, lane);
  __syncthreads();
  int hf = lane >> 4, n = lane & 15;
  // GEMM1: (16x64)@rad_W1 + b1
  for (int nt = 0; nt < 4; ++nt) {
    v8f acc = {};
#pragma unroll
    for (int ks = 0; ks < 2; ++ks)
      acc = wmma32(frag_a_lds(A, lane, ks * 32), frag_b_sw(wh + W1_H, nt, ks, lane), acc);
    int col = nt * 16 + n;
    float bb = b1[col];
#pragma unroll
    for (int r = 0; r < 8; ++r) Hw[((hf ? 8 : 0) + r) * 64 + col] = acc[r] + bb;
  }
  __syncthreads();
  // layernorm + silu -> A (f16)
  {
    int row = lane & 15, part = lane >> 4;
    float s = 0.f, ss = 0.f;
    const float* hr = Hw + row * 64 + part * 32;
#pragma unroll
    for (int j = 0; j < 32; j += 4) {
      float4 hv = *(const float4*)(hr + j);
      s  += hv.x + hv.y + hv.z + hv.w;
      ss += hv.x * hv.x + hv.y * hv.y + hv.z * hv.z + hv.w * hv.w;
    }
    s  += __shfl_xor(s, 16, 32);
    ss += __shfl_xor(ss, 16, 32);
    float mu = s * (1.0f / 64.0f);
    float var = ss * (1.0f / 64.0f) - mu * mu;
    float inv = rsqrtf(var + 1e-5f);
#pragma unroll
    for (int j = 0; j < 32; j += 4) {
      int c = part * 32 + j;
      float4 hv = *(const float4*)(hr + j);
      float4 gv = *(const float4*)(ln_g + c);
      float4 bv = *(const float4*)(ln_b + c);
      v4h o;
      o[0] = (_Float16)siluf((hv.x - mu) * inv * gv.x + bv.x);
      o[1] = (_Float16)siluf((hv.y - mu) * inv * gv.y + bv.y);
      o[2] = (_Float16)siluf((hv.z - mu) * inv * gv.z + bv.z);
      o[3] = (_Float16)siluf((hv.w - mu) * inv * gv.w + bv.w);
      *(v4h*)(A + row * 64 + c) = o;
    }
  }
  __syncthreads();
  // GEMM2: (16x64)@rad_W2 + offset -> wlc ; keep l=0 slab (cols 0-63) in LDS
  for (int nt = 0; nt < 12; ++nt) {
    v8f acc = {};
#pragma unroll
    for (int ks = 0; ks < 2; ++ks)
      acc = wmma32(frag_a_lds(A, lane, ks * 32), frag_b_sw(wh + W2_H, nt, ks, lane), acc);
    int col = nt * 16 + n;
    float ro = rad_off[col];
#pragma unroll
    for (int r = 0; r < 8; ++r) {
      int er = (hf ? 8 : 0) + r;
      float v = acc[r] + ro;
      wlc[(size_t)(e0 + er) * 192 + col] = v;
      if (nt < 4) Hw[er * 64 + col] = v;
    }
  }
  __syncthreads();
  // msg (m=0) = (lin_src + lin_dst) * edge_attr[:,0] * wlc[:,0,:] -> A (f16), vectorized
#pragma unroll
  for (int it = 0; it < 8; ++it) {
    int v = it * 32 + lane;
    int ei = v >> 4, c4 = (v & 15) * 4;
    int sn = SI[wid][ei], dn = DI[wid][ei];
    float4 a4 = *(const float4*)(lin + (size_t)(sn * 9) * 128 + c4);
    float4 b4 = *(const float4*)(lin + (size_t)(dn * 9) * 128 + 64 + c4);
    float4 w4 = *(const float4*)(Hw + ei * 64 + c4);
    float ea = EA[wid][ei];
    v4h h;
    h[0] = (_Float16)((a4.x + b4.x) * ea * w4.x);
    h[1] = (_Float16)((a4.y + b4.y) * ea * w4.y);
    h[2] = (_Float16)((a4.z + b4.z) * ea * w4.z);
    h[3] = (_Float16)((a4.w + b4.w) * ea * w4.w);
    *(v4h*)(A + ei * 64 + c4) = h;
  }
  __syncthreads();
  // alpha GEMM + b_alpha + smooth_lrelu -> Hw
  for (int nt = 0; nt < 4; ++nt) {
    v8f acc = {};
#pragma unroll
    for (int ks = 0; ks < 2; ++ks)
      acc = wmma32(frag_a_lds(A, lane, ks * 32), frag_b_sw(wh + WA_H, nt, ks, lane), acc);
    int col = nt * 16 + n;
    float ba = b_alpha[col];
#pragma unroll
    for (int r = 0; r < 8; ++r)
      Hw[((hf ? 8 : 0) + r) * 64 + col] = slrelu(acc[r] + ba);
  }
  __syncthreads();
  // per-head dot with alpha_dot -> logits
  for (int t = lane; t < 128; t += 32) {
    int ei = t >> 3, h = t & 7;
    float s = 0.f;
#pragma unroll
    for (int d = 0; d < 8; ++d) s += Hw[ei * 64 + h * 8 + d] * ADOT[h * 8 + d];
    logits[(size_t)(e0 + ei) * 8 + h] = s;
  }
}

// ---------------- K3: segment softmax plumbing ----------------
__global__ void k_init(unsigned* nmax, float* nsum, float* accum) {
  size_t t = (size_t)blockIdx.x * blockDim.x + threadIdx.x;
  size_t stride = (size_t)gridDim.x * blockDim.x;
  for (size_t i = t; i < (size_t)NN * 8; i += stride) { nmax[i] = 0u; nsum[i] = 0.f; }
  for (size_t i = t; i < (size_t)RROWS * 64; i += stride) accum[i] = 0.f;
}
__global__ void k_segmax(const float* logits, const int* edst, unsigned* nmax) {
  int t = blockIdx.x * blockDim.x + threadIdx.x;
  if (t >= NE * 8) return;
  int e = t >> 3, h = t & 7;
  atomicMax(&nmax[edst[e] * 8 + h], ford(logits[t]));
}
__global__ void k_segexp(float* logits, const int* edst, const unsigned* nmax, float* nsum) {
  int t = blockIdx.x * blockDim.x + threadIdx.x;
  if (t >= NE * 8) return;
  int e = t >> 3, h = t & 7;
  float m = funord(nmax[edst[e] * 8 + h]);
  float ex = __expf(logits[t] - m);
  logits[t] = ex;
  atomicAdd(&nsum[edst[e] * 8 + h], ex);
}

// ---------------- K4: value GEMM + gate + cp_tp + alpha-weighted scatter ----------------
__global__ void __launch_bounds__(128) k_value_scatter(
    const float* lin, const float* edge_attr, const float* wlc,
    const float* b_value, const float* elog, const float* nsum,
    const int* esrc, const int* edst, const _Float16* wh, float* accum) {
  __shared__ _Float16 As[4][1024];
  __shared__ float    Hs[4][1024];
  __shared__ float    Gs[4][1024];
  __shared__ float    AL[4][128];
  __shared__ int      SI[4][16];
  __shared__ int      DI[4][16];
  __shared__ float    EA[4][16];
  int wid = threadIdx.x >> 5, lane = threadIdx.x & 31;
  int tile = blockIdx.x * 4 + wid;      // 10000 tiles exactly
  int e0 = tile * 16;
  _Float16* A = As[wid];
  float* Hw = Hs[wid];
  float* G  = Gs[wid];
  if (lane < 16) {
    SI[wid][lane] = esrc[e0 + lane];
    DI[wid][lane] = edst[e0 + lane];
  }
  __syncthreads();
  // per-edge softmax weights, computed once per tile (independent of m)
  for (int t = lane; t < 128; t += 32) {
    int ei = t >> 3, h = t & 7;
    AL[wid][t] = elog[(size_t)(e0 + ei) * 8 + h] / (nsum[DI[wid][ei] * 8 + h] + 1e-16f);
  }
  __syncthreads();
  int hf = lane >> 4, n = lane & 15;
  for (int m = 0; m < 9; ++m) {
    int l = (m == 0) ? 0 : (m < 4 ? 1 : 2);
    if (lane < 16) EA[wid][lane] = edge_attr[(e0 + lane) * 9 + m];
    __syncthreads();
    // msg(m) tile -> A (f16), vectorized
#pragma unroll
    for (int it = 0; it < 8; ++it) {
      int v = it * 32 + lane;
      int ei = v >> 4, c4 = (v & 15) * 4;
      int e = e0 + ei;
      int sn = SI[wid][ei], dn = DI[wid][ei];
      float4 a4 = *(const float4*)(lin + (size_t)(sn * 9 + m) * 128 + c4);
      float4 b4 = *(const float4*)(lin + (size_t)(dn * 9 + m) * 128 + 64 + c4);
      float4 w4 = *(const float4*)(wlc + (size_t)e * 192 + l * 64 + c4);
      float ea = EA[wid][ei];
      v4h h;
      h[0] = (_Float16)((a4.x + b4.x) * ea * w4.x);
      h[1] = (_Float16)((a4.y + b4.y) * ea * w4.y);
      h[2] = (_Float16)((a4.z + b4.z) * ea * w4.z);
      h[3] = (_Float16)((a4.w + b4.w) * ea * w4.w);
      *(v4h*)(A + ei * 64 + c4) = h;
    }
    __syncthreads();
    // value GEMM: cols 0-63 always; cols 64-127 (gate input) only at m==0
    int ntmax = (m == 0) ? 8 : 4;
    for (int nt = 0; nt < ntmax; ++nt) {
      v8f acc = {};
#pragma unroll
      for (int ks = 0; ks < 2; ++ks)
        acc = wmma32(frag_a_lds(A, lane, ks * 32), frag_b_sw(wh + WV_H, nt, ks, lane), acc);
      int col = nt * 16 + n;
      float bv = (nt >= 4) ? b_value[col] : 0.f;
#pragma unroll
      for (int r = 0; r < 8; ++r) {
        int er = (hf ? 8 : 0) + r;
        if (nt < 4) Hw[er * 64 + col] = acc[r];
        else        G[er * 64 + (col - 64)] = sigmf(acc[r] + bv);   // persists over m
      }
    }
    __syncthreads();
    // gate, second cp_tp, alpha weight, atomic scatter by edge_dst
#pragma unroll
    for (int it = 0; it < 8; ++it) {
      int v = it * 32 + lane;
      int ei = v >> 4, c4 = (v & 15) * 4;
      int e = e0 + ei;
      int dn = DI[wid][ei];
      float4 hv = *(const float4*)(Hw + ei * 64 + c4);
      float4 wv = *(const float4*)(wlc + (size_t)e * 192 + l * 64 + c4);
      float ea = EA[wid][ei];
      float x[4] = {hv.x, hv.y, hv.z, hv.w};
      float w4[4] = {wv.x, wv.y, wv.z, wv.w};
      if (m == 0) {
        float4 bv = *(const float4*)(b_value + c4);
        x[0] = siluf(x[0] + bv.x); x[1] = siluf(x[1] + bv.y);
        x[2] = siluf(x[2] + bv.z); x[3] = siluf(x[3] + bv.w);
      } else {
        float4 gv = *(const float4*)(G + ei * 64 + c4);
        x[0] *= gv.x; x[1] *= gv.y; x[2] *= gv.z; x[3] *= gv.w;
      }
      float* accr = accum + (size_t)(dn * 9 + m) * 64 + c4;
      float al = AL[wid][ei * 8 + (c4 >> 3)];   // head = c>>3 constant across the 4-vec
#pragma unroll
      for (int q = 0; q < 4; ++q)
        atomicAdd(accr + q, x[q] * ea * w4[q] * al);
    }
    __syncthreads();
  }
}

// ---------------- K5: output projection (no bias; fixup kernel adds b_proj) ----------------
__global__ void __launch_bounds__(256) k_proj(const float* accum, const _Float16* wh,
                                              float* out) {
  __shared__ _Float16 As[8][1024];
  int wid = threadIdx.x >> 5, lane = threadIdx.x & 31;
  int tile = blockIdx.x * 8 + wid;
  bool active = tile * 16 < RROWS;
  int row0 = tile * 16;
  _Float16* A = As[wid];
  if (active) stage_a16x64(accum + (size_t)row0 * 64, A, lane);
  __syncthreads();
  int hf = lane >> 4, n = lane & 15;
  for (int nt = 0; nt < 4; ++nt) {
    v8f acc = {};
#pragma unroll
    for (int ks = 0; ks < 2; ++ks)
      acc = wmma32(frag_a_lds(A, lane, ks * 32), frag_b_sw(wh + WP_H, nt, ks, lane), acc);
    if (active) {
      int col = nt * 16 + n;
      float* outp = out + (size_t)(row0 + (hf ? 8 : 0)) * 64 + col;
#pragma unroll
      for (int r = 0; r < 8; ++r) outp[r * 64] = acc[r];
    }
  }
}

extern "C" void kernel_launch(void* const* d_in, const int* in_sizes, int n_in,
                              void* d_out, int out_size, void* d_ws, size_t ws_size,
                              hipStream_t stream) {
  const float* node_input = (const float*)d_in[0];
  const float* edge_attr  = (const float*)d_in[1];
  const float* edge_scal  = (const float*)d_in[2];
  const float* W_src      = (const float*)d_in[3];
  const float* b_src      = (const float*)d_in[4];
  const float* W_dst      = (const float*)d_in[5];
  const float* rad_W1     = (const float*)d_in[6];
  const float* rad_b1     = (const float*)d_in[7];
  const float* ln_g       = (const float*)d_in[8];
  const float* ln_b       = (const float*)d_in[9];
  const float* rad_W2     = (const float*)d_in[10];
  const float* rad_off    = (const float*)d_in[11];
  const float* W_alpha    = (const float*)d_in[12];
  const float* b_alpha    = (const float*)d_in[13];
  const float* alpha_dot  = (const float*)d_in[14];
  const float* W_value    = (const float*)d_in[15];
  const float* b_value    = (const float*)d_in[16];
  const float* W_proj     = (const float*)d_in[17];
  const float* b_proj     = (const float*)d_in[18];
  const int*   esrc       = (const int*)d_in[19];
  const int*   edst       = (const int*)d_in[20];

  char* ws = (char*)d_ws;
  _Float16* wh   = (_Float16*)(ws + OFF_WH);
  float* lin     = (float*)(ws + OFF_LIN);
  float* wlc     = (float*)(ws + OFF_WLC);
  float* logits  = (float*)(ws + OFF_LOG);
  unsigned* nmax = (unsigned*)(ws + OFF_NMAX);
  float* nsum    = (float*)(ws + OFF_NSUM);
  float* accum   = (float*)(ws + OFF_ACC);

  int biasGrid = (NN * 64 + 255) / 256;
  k_convert_weights<<<40, 256, 0, stream>>>(W_src, W_dst, rad_W1, rad_W2, W_alpha, W_value,
                                            W_proj, wh);
  k_node_lin<<<704, 256, 0, stream>>>(node_input, wh, lin);                 // 5625 tiles
  k_rowbias<<<biasGrid, 256, 0, stream>>>(lin, b_src, 128);                 // b_src on m==0 rows
  k_radial_alpha<<<1250, 256, 0, stream>>>(edge_scal, rad_b1, ln_g, ln_b, rad_off, b_alpha,
                                           alpha_dot, edge_attr, esrc, edst, lin, wh, wlc,
                                           logits);                          // 10000 tiles
  k_init<<<512, 256, 0, stream>>>(nmax, nsum, accum);
  k_segmax<<<(NE * 8 + 255) / 256, 256, 0, stream>>>(logits, edst, nmax);
  k_segexp<<<(NE * 8 + 255) / 256, 256, 0, stream>>>(logits, edst, nmax, nsum);
  k_value_scatter<<<2500, 128, 0, stream>>>(lin, edge_attr, wlc, b_value, logits, nsum,
                                            esrc, edst, wh, accum);          // 10000 tiles
  k_proj<<<704, 256, 0, stream>>>(accum, wh, (float*)d_out);
  k_rowbias<<<biasGrid, 256, 0, stream>>>((float*)d_out, b_proj, 64);        // b_proj on m==0 rows
}
